// Attention_17334488007364
// MI455X (gfx1250) — matrix operand order repaired and checked by reading, not compiled, not run on hardware
//
#include <hip/hip_runtime.h>

typedef __attribute__((ext_vector_type(16))) _Float16 v16h;
typedef __attribute__((ext_vector_type(8)))  float    v8f;

#define B_   16
#define C_   128
#define N_   4096   // 64*64
#define M_   1024   // 32*32
#define LDS_STRIDE 1028  // 1024 + 4 dword pad -> conflict-free row reads

// 16-bit A-matrix 16x32 element->K map (contiguous, matches B fragments):
// lanes 0-15 (h=0): e0..15 -> K0..15
// lanes 16-31(h=1): e0..15 -> K16..31
__device__ __forceinline__ int kmapA(int e, int h) {
    return 16 * h + e;
}

__device__ __forceinline__ v8f wmma_f16(v16h a, v16h b, v8f c) {
    return __builtin_amdgcn_wmma_f32_16x16x32_f16(false, a, false, b, (short)0, c, false, false);
}

// ---------------- Kernel 1: fused QKV projection ----------------
// D(oc, n) = W_all(96 x 128) * x(128 x n), per batch. One wave = one 16-col n-tile,
// computing all 6 oc-tiles (x B-fragment reused 6x). q -> f16 [b][n][16] (A-layout
// friendly for scores); k/v -> f32 temp [b][80][n] for pooling.
__global__ void proj_kernel(const float* __restrict__ x,
                            const float* __restrict__ wq,
                            const float* __restrict__ wk,
                            const float* __restrict__ wv,
                            _Float16* __restrict__ qbuf,
                            float* __restrict__ kvfull)
{
    const int lane = threadIdx.x & 31;
    const int wid  = blockIdx.x * (blockDim.x >> 5) + (threadIdx.x >> 5);
    const int b    = wid >> 8;            // 256 n-tiles per batch
    const int n0   = (wid & 255) << 4;
    const int h    = lane >> 4;
    const int r    = lane & 15;

    v8f acc[6];
#pragma unroll
    for (int t = 0; t < 6; ++t) acc[t] = (v8f){};

    const float* xb = x + (size_t)b * C_ * N_;

    for (int kc = 0; kc < 4; ++kc) {      // K = 128 in 4 chunks of 32
        const int c0 = kc * 32;
        v16h bx;                           // B-frag: x[c, n]
#pragma unroll
        for (int e = 0; e < 16; ++e) {
            int c = c0 + 16 * h + e;
            bx[e] = (_Float16)xb[(size_t)c * N_ + n0 + r];
        }
#pragma unroll
        for (int t = 0; t < 6; ++t) {
            const int oc = t * 16 + r;     // A row = stacked weight row
            const float* wrow = (oc < 16) ? (wq + oc * C_)
                              : (oc < 32) ? (wk + (oc - 16) * C_)
                                          : (wv + (oc - 32) * C_);
            v16h aw;
#pragma unroll
            for (int e = 0; e < 16; ++e)
                aw[e] = (_Float16)wrow[c0 + kmapA(e, h)];
            acc[t] = wmma_f16(aw, bx, acc[t]);
        }
    }
#pragma unroll
    for (int t = 0; t < 6; ++t) {
#pragma unroll
        for (int g = 0; g < 8; ++g) {      // C-layout: row oc = t*16+g+8h, col n = n0+r
            int oc = t * 16 + g + 8 * h;
            int n  = n0 + r;
            float v = acc[t][g];
            if (oc < 16)
                qbuf[((size_t)b * N_ + n) * 16 + oc] = (_Float16)v;
            else
                kvfull[((size_t)b * 80 + (oc - 16)) * N_ + n] = v;
        }
    }
}

// ---------------- Kernel 2: 2x2 max-pool k/v, f32 -> f16 ----------------
// k stored TRANSPOSED [b][m][16ch] so the scores B-fragment is one contiguous
// 32B load per lane; v stored [b][64ch][m] (contiguous over m = PV B-fragment).
__global__ void pool_kernel(const float* __restrict__ kvfull,
                            _Float16* __restrict__ kpoolT,
                            _Float16* __restrict__ vpool)
{
    int idx = blockIdx.x * blockDim.x + threadIdx.x;   // B*80*M
    if (idx >= B_ * 80 * M_) return;
    int m  = idx & (M_ - 1);
    int oc = (idx >> 10) % 80;
    int b  = idx / (80 * M_);
    int h2 = m >> 5, w2 = m & 31;
    const float* s = kvfull + ((size_t)b * 80 + oc) * N_ + (h2 * 2) * 64 + w2 * 2;
    float mx = fmaxf(fmaxf(s[0], s[1]), fmaxf(s[64], s[65]));
    if (oc < 16) kpoolT[((size_t)b * M_ + m) * 16 + oc] = (_Float16)mx;
    else         vpool[((size_t)b * 64 + (oc - 16)) * M_ + m] = (_Float16)mx;
}

// ---------------- Kernel 3: attention (scores + softmax + PV) ----------------
// One wave = one (b, 16-row q-tile). Pass A: 64 WMMAs -> S[16][1024] in a private
// LDS strip (padded stride), tracking row-max. Pass B: rebuild P = exp(S-max) into
// A-fragments, 128 WMMAs vs v, row-sums via shfl_xor(16), normalize.
__global__ void attn_kernel(const _Float16* __restrict__ qbuf,
                            const _Float16* __restrict__ kpoolT,
                            const _Float16* __restrict__ vpool,
                            _Float16* __restrict__ obuf)
{
    extern __shared__ float smem[];
    const int lane  = threadIdx.x & 31;
    const int wslot = threadIdx.x >> 5;
    const int wid   = blockIdx.x * (blockDim.x >> 5) + wslot;
    const int b     = wid >> 8;
    const int n0    = (wid & 255) << 4;
    const int h     = lane >> 4;
    const int r     = lane & 15;
    float* S = smem + (size_t)wslot * 16 * LDS_STRIDE;   // private per-wave strip

    // Q A-fragment: K = 16 channels, zero-padded to 32 (matching zero pad on B side)
    v16h aq;
#pragma unroll
    for (int e = 0; e < 16; ++e) {
        int ch = kmapA(e, h);
        aq[e] = (ch < 16) ? qbuf[((size_t)b * N_ + n0 + r) * 16 + ch] : (_Float16)0.f;
    }

    const _Float16* kbT = kpoolT + (size_t)b * M_ * 16;
    float mloc[8];
#pragma unroll
    for (int g = 0; g < 8; ++g) mloc[g] = -3.0e38f;

    for (int mt = 0; mt < 64; ++mt) {          // Pass A: scores
        const int m0 = mt * 16;
        // B-frag: k[ch, m] = kpoolT[m][ch]; one contiguous 32B load per lane (h==0),
        // lanes 16-31 carry the zero-padded K=16..31 rows.
        v16h bk = (v16h){};
        if (h == 0) bk = *(const v16h*)(kbT + (size_t)(m0 + r) * 16);
        if (mt + 4 < 64)                        // stream next k tiles through L2
            __builtin_prefetch(kbT + (size_t)(m0 + 64 + r) * 16, 0, 1);
        v8f z = (v8f){};
        v8f s = wmma_f16(aq, bk, z);
#pragma unroll
        for (int g = 0; g < 8; ++g) {
            S[(g + 8 * h) * LDS_STRIDE + m0 + r] = s[g];
            mloc[g] = fmaxf(mloc[g], s[g]);
        }
    }
    // Row-max: reduce across the 16 lanes of each half (rows 8h..8h+7)
#pragma unroll
    for (int g = 0; g < 8; ++g) {
        mloc[g] = fmaxf(mloc[g], __shfl_xor(mloc[g], 1, 32));
        mloc[g] = fmaxf(mloc[g], __shfl_xor(mloc[g], 2, 32));
        mloc[g] = fmaxf(mloc[g], __shfl_xor(mloc[g], 4, 32));
        mloc[g] = fmaxf(mloc[g], __shfl_xor(mloc[g], 8, 32));
    }
    // Broadcast max of this lane's A-row r (rows 0-7 live in lanes 0-15, 8-15 in 16-31)
    float mrow = 0.f;
    const int srcLane = (r >> 3) << 4;
#pragma unroll
    for (int g = 0; g < 8; ++g) {
        float t = __shfl(mloc[g], srcLane, 32);
        if ((r & 7) == g) mrow = t;
    }

    v8f oacc[4];
#pragma unroll
    for (int ct = 0; ct < 4; ++ct) oacc[ct] = (v8f){};
    float lsum = 0.f;
    const _Float16* vbh = vpool + (size_t)b * 64 * M_;

    for (int mc = 0; mc < 32; ++mc) {          // Pass B: P*V, K = 32 m per step
        const int m0 = mc * 32;
        v16h ap;                                // A-frag of P, exp on the fly
#pragma unroll
        for (int e = 0; e < 16; ++e) {
            int m = m0 + kmapA(e, h);
            float p = __expf(S[r * LDS_STRIDE + m] - mrow);
            lsum += p;
            ap[e] = (_Float16)p;
        }
        if (mc + 2 < 32)                        // stream next v chunk through L2
            __builtin_prefetch(vbh + (size_t)r * M_ + m0 + 64, 0, 1);
#pragma unroll
        for (int ct = 0; ct < 4; ++ct) {
            int c = ct * 16 + r;                // B-frag: v[c][m], 32B contiguous per lane
            const v16h* vb = (const v16h*)(vbh + (size_t)c * M_ + m0 + 16 * h);
            oacc[ct] = wmma_f16(ap, vb[0], oacc[ct]);
        }
    }
    lsum += __shfl_xor(lsum, 16, 32);           // combine half-K partials of row r
#pragma unroll
    for (int ct = 0; ct < 4; ++ct) {
#pragma unroll
        for (int g = 0; g < 8; ++g) {
            float l = __shfl(lsum, g + 8 * h, 32);  // lane index == row index
            int n = n0 + g + 8 * h;
            obuf[((size_t)b * N_ + n) * 64 + ct * 16 + r] = (_Float16)(oacc[ct][g] / l);
        }
    }
}

// ---------------- Kernel 4: output projection + residual ----------------
// D(co, n) = wo(128x64) * o(64 x n); A = wo so D cols = n -> coalesced f32 stores.
__global__ void outproj_kernel(const _Float16* __restrict__ obuf,
                               const float* __restrict__ wo,
                               const float* __restrict__ x,
                               const float* __restrict__ gamma,
                               float* __restrict__ out)
{
    const int lane = threadIdx.x & 31;
    const int wid  = blockIdx.x * (blockDim.x >> 5) + (threadIdx.x >> 5);
    const int b    = wid >> 8;
    const int n0   = (wid & 255) << 4;
    const int h    = lane >> 4;
    const int r    = lane & 15;
    const float gm = gamma[0];

    v16h bo[2];                                 // B-frag: o[c][n] = obuf[b][n][c]
    const _Float16* ob = obuf + ((size_t)b * N_ + n0 + r) * 64;
#pragma unroll
    for (int kc = 0; kc < 2; ++kc)
        bo[kc] = *(const v16h*)(ob + kc * 32 + 16 * h);

#pragma unroll
    for (int rt = 0; rt < 8; ++rt) {
        const float* wrow = wo + (rt * 16 + r) * 64;  // A row co = rt*16 + r
        v8f acc = (v8f){};
#pragma unroll
        for (int kc = 0; kc < 2; ++kc) {
            v16h aw;
#pragma unroll
            for (int e = 0; e < 16; ++e)
                aw[e] = (_Float16)wrow[kc * 32 + kmapA(e, h)];
            acc = wmma_f16(aw, bo[kc], acc);
        }
#pragma unroll
        for (int g = 0; g < 8; ++g) {
            int co = rt * 16 + g + 8 * h;
            size_t off = ((size_t)b * C_ + co) * N_ + n0 + r;
            out[off] = gm * acc[g] + x[off];
        }
    }
}

extern "C" void kernel_launch(void* const* d_in, const int* in_sizes, int n_in,
                              void* d_out, int out_size, void* d_ws, size_t ws_size,
                              hipStream_t stream) {
    const float* x     = (const float*)d_in[0];
    const float* wq    = (const float*)d_in[1];
    const float* wk    = (const float*)d_in[2];
    const float* wv    = (const float*)d_in[3];
    const float* wo    = (const float*)d_in[4];
    const float* gamma = (const float*)d_in[5];
    float* out = (float*)d_out;

    char* ws = (char*)d_ws;
    _Float16* qbuf   = (_Float16*)(ws);                      //  2,097,152 B
    float*    kvfull = (float*)   (ws + 2097152u);           // 20,971,520 B
    _Float16* kpoolT = (_Float16*)(ws + 23068672u);          //    524,288 B
    _Float16* vpool  = (_Float16*)(ws + 23592960u);          //  2,097,152 B
    _Float16* obuf   = (_Float16*)(ws + 25690112u);          //  8,388,608 B (end 34,078,720)

    const int tiles = B_ * (N_ / 16);                        // 4096 wave-tiles
    proj_kernel<<<tiles / 4, 128, 0, stream>>>(x, wq, wk, wv, qbuf, kvfull);
    pool_kernel<<<(B_ * 80 * M_ + 255) / 256, 256, 0, stream>>>(kvfull, kpoolT, vpool);
    attn_kernel<<<tiles / 2, 64, 2 * 16 * LDS_STRIDE * sizeof(float), stream>>>(
        qbuf, kpoolT, vpool, obuf);
    outproj_kernel<<<tiles / 4, 128, 0, stream>>>(obuf, wo, x, gamma, out);
}
